// PooledSelfAttention2d_3358664425661
// MI455X (gfx1250) — matrix operand order repaired and checked by reading, not compiled, not run on hardware
//
#include <hip/hip_runtime.h>

// ---------------------------------------------------------------------------
// PooledSelfAttention2d for MI455X (gfx1250, wave32)
//   B=16, C=64, H=W=64, HW=4096, pooled HWp=1024, Cq=C/8=8, Cv=C/2=32
//
// K1: proj+pool  -> f_ws [B][8][4096] f32, g_ws [B][8][1024] f32,
//                   hfrag [B][32 mchunks][2 ctiles][32 lanes][8 dw] bf16x2
//                   (hT pre-swizzled into the WMMA bf16 B-operand layout)
// K2: flash attn -> o_ws [B][4096][32] f32   (O^T, n-major)
//       S^T tile = g^T x f via V_WMMA_F32_16X16X4_F32 (K=8 -> 2 wmma)
//       P(bf16) x h^T via V_WMMA_F32_16X16X32_BF16, online softmax
//       g staged to LDS via GLOBAL_LOAD_ASYNC_TO_LDS_B128
// K3: out = gamma * (w_o x o) + x
// ---------------------------------------------------------------------------

typedef __attribute__((ext_vector_type(2)))  float  v2f;
typedef __attribute__((ext_vector_type(8)))  float  v8f;
typedef __attribute__((ext_vector_type(4)))  int    v4i;
typedef __attribute__((ext_vector_type(16))) __bf16 v16bf;

typedef __attribute__((address_space(1))) v4i gas_v4i;   // global int4
typedef __attribute__((address_space(3))) v4i las_v4i;   // LDS int4

#define B_    16
#define C_    64
#define HW_   4096
#define HWP_  1024

#if defined(__gfx1250__) && __has_builtin(__builtin_amdgcn_global_load_async_to_lds_b128)
#define HAVE_ASYNC_LDS 1
#else
#define HAVE_ASYNC_LDS 0
#endif

__device__ __forceinline__ unsigned short f2bf(float x) {
  union { float f; unsigned u; } v; v.f = x;
  unsigned r = v.u + 0x7FFFu + ((v.u >> 16) & 1u);   // RNE
  return (unsigned short)(r >> 16);
}

__device__ __forceinline__ unsigned pack2_bf16(float lo, float hi) {
  union { float f; unsigned u; } a, b; a.f = lo; b.f = hi;
  unsigned ul = a.u + 0x7FFFu + ((a.u >> 16) & 1u);
  unsigned uh = b.u + 0x7FFFu + ((b.u >> 16) & 1u);
  return (ul >> 16) | (uh & 0xFFFF0000u);
}

// ---------------------------------------------------------------------------
// Kernel 1: fused 1x1 conv projections (theta/phi/g) + 2x2 maxpool for phi/g.
// One thread per SOURCE position; lanes {t,t^1,t^2,t^3} form one pool cell,
// reduced with two shfl_xor.  256 blocks -> fills the chip.
// ---------------------------------------------------------------------------
__global__ __launch_bounds__(256) void proj_pool_kernel(
    const float* __restrict__ x, const float* __restrict__ w_theta,
    const float* __restrict__ w_phi, const float* __restrict__ w_g,
    float* __restrict__ f_ws, float* __restrict__ g_ws,
    unsigned short* __restrict__ hfrag_bf)
{
  __shared__ float wts[48 * 64];           // rows: 0-7 theta, 8-15 phi, 16-47 g
  const int tid = threadIdx.x;
  for (int i = tid; i < 8 * 64;  i += 256) wts[i]            = w_theta[i];
  for (int i = tid; i < 8 * 64;  i += 256) wts[8 * 64 + i]   = w_phi[i];
  for (int i = tid; i < 32 * 64; i += 256) wts[16 * 64 + i]  = w_g[i];
  __syncthreads();

  const int b    = blockIdx.x;
  const int cell = blockIdx.y * 64 + (tid >> 2);     // pooled cell 0..1023
  const int hp = cell >> 5, wp = cell & 31;
  const int dh = (tid >> 1) & 1, dw = tid & 1;
  const int n  = (2 * hp + dh) * 64 + (2 * wp + dw); // source position

  const float* xb = x + (size_t)b * C_ * HW_;

  float acc[48];
  #pragma unroll
  for (int o = 0; o < 48; ++o) acc[o] = 0.f;
  for (int c0 = 0; c0 < 64; c0 += 8) {
    float xv[8];
    #pragma unroll
    for (int j = 0; j < 8; ++j) xv[j] = xb[(size_t)(c0 + j) * HW_ + n];
    #pragma unroll
    for (int o = 0; o < 48; ++o) {
      #pragma unroll
      for (int j = 0; j < 8; ++j) acc[o] += wts[o * 64 + c0 + j] * xv[j];
    }
  }

  // theta projection: store every source position
  #pragma unroll
  for (int o = 0; o < 8; ++o)
    f_ws[((size_t)b * 8 + o) * HW_ + n] = acc[o];

  // 2x2 max pool across the 4 sub-lanes of this cell
  #pragma unroll
  for (int o = 8; o < 48; ++o) {
    float v = acc[o];
    v = fmaxf(v, __shfl_xor(v, 1, 32));
    v = fmaxf(v, __shfl_xor(v, 2, 32));
    acc[o] = v;
  }

  if ((tid & 3) == 0) {
    #pragma unroll
    for (int o = 0; o < 8; ++o)
      g_ws[((size_t)b * 8 + o) * HWP_ + cell] = acc[8 + o];

    // Scatter h^T into the bf16 WMMA B-operand (32x16) register layout:
    //   lane = n_local + ((k>>3)&1)*16 ; reg r = ((k&7) + (k>>4)*8)>>1 ; packed slot p = k&1
    const int mc = cell >> 5, k = cell & 31;
    const int lhalf = ((k >> 3) & 1) ? 16 : 0;
    const int rr = ((k & 7) + ((k >> 4) << 3)) >> 1, pp = k & 1;
    #pragma unroll
    for (int c = 0; c < 32; ++c) {
      const int ct = c >> 4, nl = c & 15;
      size_t dword = (((size_t)b * 32 + mc) * 2 + ct) * 256
                   + (size_t)(nl + lhalf) * 8 + rr;
      hfrag_bf[dword * 2 + pp] = f2bf(acc[16 + c]);
    }
  }
}

// ---------------------------------------------------------------------------
// Kernel 2: flash attention. One wave per 16-query tile; 8 waves/block share
// one batch's g (staged in LDS). m processed in chunks of 32.
// ---------------------------------------------------------------------------
__global__ __launch_bounds__(256) void attn_kernel(
    const float* __restrict__ f_ws, const float* __restrict__ g_ws,
    const unsigned* __restrict__ hfrag, float* __restrict__ o_ws)
{
  __shared__ float gsh[8 * HWP_];          // 32 KB: g for this batch
  const int tid = threadIdx.x;
  const int b   = blockIdx.x;

#if HAVE_ASYNC_LDS
  {
    const float* gsrc = g_ws + (size_t)b * 8 * HWP_;
    for (int i = tid; i < (8 * HWP_) / 4; i += 256) {
      __builtin_amdgcn_global_load_async_to_lds_b128(
          (gas_v4i*)(gsrc + 4 * i),
          (las_v4i*)(&gsh[4 * i]),
          0, 0);
    }
#if __has_builtin(__builtin_amdgcn_s_wait_asynccnt)
    __builtin_amdgcn_s_wait_asynccnt(0);
#else
    asm volatile("s_wait_asynccnt 0x0" ::: "memory");
#endif
  }
#else
  for (int i = tid; i < 8 * HWP_; i += 256) gsh[i] = g_ws[(size_t)b * 8 * HWP_ + i];
#endif
  __syncthreads();

  const int wave = tid >> 5;
  const int lane = tid & 31;
  const int n0   = (blockIdx.y * 8 + wave) * 16;   // query tile base
  const int ln   = lane & 15;
  const int hsel = (lane >= 16) ? 1 : 0;

  // B-fragment of scores GEMM (f, K=c, N=n), loaded once per tile.
  const float* fb = f_ws + (size_t)b * 8 * HW_;
  v2f fB0, fB1;
  fB0.x = fb[(size_t)(0 + 2 * hsel) * HW_ + n0 + ln];
  fB0.y = fb[(size_t)(1 + 2 * hsel) * HW_ + n0 + ln];
  fB1.x = fb[(size_t)(4 + 2 * hsel) * HW_ + n0 + ln];
  fB1.y = fb[(size_t)(5 + 2 * hsel) * HW_ + n0 + ln];

  v8f acc0 = {}; v8f acc1 = {};
  float runmax = -3.0e38f, runsum = 0.f;

  const unsigned* hblk = hfrag + (size_t)b * 32 * 2 * 256;

  for (int mc = 0; mc < 32; ++mc) {
    if (mc + 1 < 32)
      __builtin_prefetch(hblk + (size_t)(mc + 1) * 2 * 256 + lane * 8, 0, 1);

    // ---- scores S^T for two 16-m tiles (fp32 WMMA, K=8 = 2x K=4) ----
    v8f s0 = {}; v8f s1 = {};
    {
      int m = mc * 32 + ln;
      v2f a0 = { gsh[(2 * hsel + 0) * HWP_ + m], gsh[(2 * hsel + 1) * HWP_ + m] };
      v2f a1 = { gsh[(2 * hsel + 4) * HWP_ + m], gsh[(2 * hsel + 5) * HWP_ + m] };
      s0 = __builtin_amdgcn_wmma_f32_16x16x4_f32(false, a0, false, fB0, (short)0, s0, false, false);
      s0 = __builtin_amdgcn_wmma_f32_16x16x4_f32(false, a1, false, fB1, (short)0, s0, false, false);
      m += 16;
      v2f a2 = { gsh[(2 * hsel + 0) * HWP_ + m], gsh[(2 * hsel + 1) * HWP_ + m] };
      v2f a3 = { gsh[(2 * hsel + 4) * HWP_ + m], gsh[(2 * hsel + 5) * HWP_ + m] };
      s1 = __builtin_amdgcn_wmma_f32_16x16x4_f32(false, a2, false, fB0, (short)0, s1, false, false);
      s1 = __builtin_amdgcn_wmma_f32_16x16x4_f32(false, a3, false, fB1, (short)0, s1, false, false);
    }

    // ---- online softmax (stats keyed per-lane: n = lane%16) ----
    float cm = -3.0e38f;
    #pragma unroll
    for (int i = 0; i < 8; ++i) { cm = fmaxf(cm, s0[i]); cm = fmaxf(cm, s1[i]); }
    cm = fmaxf(cm, __shfl_xor(cm, 16, 32));
    const float newmax = fmaxf(runmax, cm);
    const float scalef = __expf(runmax - newmax);
    runmax = newmax;

    float csum = 0.f;
    #pragma unroll
    for (int i = 0; i < 8; ++i) {
      s0[i] = __expf(s0[i] - newmax);
      s1[i] = __expf(s1[i] - newmax);
      csum += s0[i] + s1[i];
    }
    csum += __shfl_xor(csum, 16, 32);
    runsum = runsum * scalef + csum;

    // rescale accumulators (per-reg n broadcast via bpermute)
    #pragma unroll
    for (int r = 0; r < 8; ++r) {
      const float sr = __shfl(scalef, r + hsel * 8, 32);
      acc0[r] *= sr; acc1[r] *= sr;
    }

    // ---- pack P: S^T D-frag -> bf16 A-frag (pure per-lane pairwise pack) ----
    union { unsigned u[8]; v16bf v; } pa;
    #pragma unroll
    for (int j = 0; j < 4; ++j) {
      pa.u[j]     = pack2_bf16(s0[2 * j], s0[2 * j + 1]);
      pa.u[4 + j] = pack2_bf16(s1[2 * j], s1[2 * j + 1]);
    }

    // ---- h^T B-fragments (pre-swizzled; 2x b128 per fragment) ----
    union { uint4 q[2]; v16bf v; } hb0, hb1;
    const uint4* hp0 = (const uint4*)(hblk + ((size_t)mc * 2 + 0) * 256 + (size_t)lane * 8);
    const uint4* hp1 = (const uint4*)(hblk + ((size_t)mc * 2 + 1) * 256 + (size_t)lane * 8);
    hb0.q[0] = hp0[0]; hb0.q[1] = hp0[1];
    hb1.q[0] = hp1[0]; hb1.q[1] = hp1[1];

    acc0 = __builtin_amdgcn_wmma_f32_16x16x32_bf16(false, pa.v, false, hb0.v, (short)0, acc0, false, false);
    acc1 = __builtin_amdgcn_wmma_f32_16x16x32_bf16(false, pa.v, false, hb1.v, (short)0, acc1, false, false);
  }

  // ---- normalize and store O^T [n][c] ----
  #pragma unroll
  for (int r = 0; r < 8; ++r) {
    const float rs  = __shfl(runsum, r + hsel * 8, 32);
    const float inv = 1.0f / rs;
    const int n = n0 + r + hsel * 8;
    o_ws[((size_t)b * HW_ + n) * 32 + 0  + ln] = acc0[r] * inv;
    o_ws[((size_t)b * HW_ + n) * 32 + 16 + ln] = acc1[r] * inv;
  }
}

// ---------------------------------------------------------------------------
// Kernel 3: out = gamma * (w_o x o) + x
// ---------------------------------------------------------------------------
__global__ __launch_bounds__(256) void out_proj_kernel(
    const float* __restrict__ x, const float* __restrict__ w_o,
    const float* __restrict__ gamma, const float* __restrict__ o_ws,
    float* __restrict__ out)
{
  __shared__ float wsh[64 * 32];
  const int tid = threadIdx.x;
  for (int i = tid; i < 64 * 32; i += 256) wsh[i] = w_o[i];
  __syncthreads();

  const int b = blockIdx.x;
  const int n = blockIdx.y * 256 + tid;
  const float gm = gamma[0];

  float ov[32];
  const float4* op = (const float4*)(o_ws + ((size_t)b * HW_ + n) * 32);
  #pragma unroll
  for (int j = 0; j < 8; ++j) {
    float4 t = op[j];
    ov[4 * j] = t.x; ov[4 * j + 1] = t.y; ov[4 * j + 2] = t.z; ov[4 * j + 3] = t.w;
  }

  const float* xb = x   + (size_t)b * C_ * HW_ + n;
  float*       ob = out + (size_t)b * C_ * HW_ + n;
  for (int oc = 0; oc < 64; ++oc) {
    float acc = 0.f;
    #pragma unroll
    for (int c = 0; c < 32; ++c) acc += wsh[oc * 32 + c] * ov[c];
    ob[(size_t)oc * HW_] = gm * acc + xb[(size_t)oc * HW_];
  }
}

// ---------------------------------------------------------------------------
extern "C" void kernel_launch(void* const* d_in, const int* in_sizes, int n_in,
                              void* d_out, int out_size, void* d_ws, size_t ws_size,
                              hipStream_t stream)
{
  const float* x       = (const float*)d_in[0];
  const float* w_theta = (const float*)d_in[1];
  const float* w_phi   = (const float*)d_in[2];
  const float* w_g     = (const float*)d_in[3];
  const float* w_o     = (const float*)d_in[4];
  const float* gamma   = (const float*)d_in[5];

  char* ws = (char*)d_ws;
  float*    f_ws  = (float*)ws;                                           // 2 MB
  float*    g_ws  = (float*)(ws + (2u << 20));                            // 512 KB
  unsigned* hfrag = (unsigned*)(ws + (2u << 20) + (512u << 10));          // 1 MB
  float*    o_ws  = (float*)(ws + (2u << 20) + (512u << 10) + (1u << 20));// 8 MB

  proj_pool_kernel<<<dim3(16, 16), 256, 0, stream>>>(
      x, w_theta, w_phi, w_g, f_ws, g_ws, (unsigned short*)hfrag);
  attn_kernel<<<dim3(16, 32), 256, 0, stream>>>(f_ws, g_ws, hfrag, o_ws);
  out_proj_kernel<<<dim3(16, 16), 256, 0, stream>>>(
      x, w_o, gamma, o_ws, (float*)d_out);
}
